// InteractionEncoder_51522427683088
// MI455X (gfx1250) — matrix operand order repaired and checked by reading, not compiled, not run on hardware
//
#include <hip/hip_runtime.h>

typedef __attribute__((ext_vector_type(16))) __bf16 bf16x16;
typedef __attribute__((ext_vector_type(8)))  __bf16 bf16x8;
typedef __attribute__((ext_vector_type(8)))  float  f32x8;

#define LN_EPS 1e-5f

// packed bf16 weight offsets inside d_ws (units: bf16 elements)
#define OFF_FE_W1 0
#define OFF_FE_W2 4096
#define OFF_EE_W1 20480
#define OFF_EE_W2 24576
#define OFF_IE_W1 40960
#define OFF_IE_W2 90112

// LDS layout (units: bf16 elements). Pitches are padded so that consecutive
// rows start at rotating LDS banks (row stride in DWORDs is != 0 mod 64),
// giving conflict-free 16-row A-fragment loads with purely affine addressing
// (one base VGPR + immediate DS offsets; no per-element swizzle math).
#define SPITCH  40   // sF/rF rows: 80B (20 DW rotation, conflict-free)
#define EPITCH  32   // eF: unpadded (single k-chunk, used once; minor conflicts OK)
#define HPITCH  136  // hid rows: 272B (68 DW == 4 mod 64, conflict-free)
#define MPITCH  392  // msg rows: 784B (196 DW == 4 mod 64, conflict-free)

#define SF_OFF   0              // [128][40]
#define RF_OFF   5120           // [128][40]
#define EF_OFF   10240          // [128][32]
#define HID_OFF  14336          // [128][136]
#define MSG_OFF  31744          // [128][392]
#define SMEM_HALVES 81920       // 163840 bytes -> exactly 2 WGs per 320KB WGP

// ---------------------------------------------------------------------------
// Weight packer: fp32 [K x 128] -> bf16 WMMA B-fragments.
// Fragment (kc,nt) is a 32x16 K-major block; lane l<16 holds K = kc*32+0..7|16..23
// of column nt*16+l, lane>=16 holds K = kc*32+8..15|24..31 (ISA 7.12.2 B layout).
// dst index == flat thread index by construction.
// ---------------------------------------------------------------------------
__global__ void pack_weights(const float* __restrict__ W, __bf16* __restrict__ dst,
                             int Kreal, int kchunks) {
  int t = blockIdx.x * blockDim.x + threadIdx.x;
  int total = kchunks * 4096;
  if (t >= total) return;
  int i    = t & 15;
  int lane = (t >> 4) & 31;
  int nt   = (t >> 9) & 7;
  int kc   = t >> 12;
  int krow = kc * 32 + ((lane >> 4) << 3) + (i & 7) + ((i >> 3) << 4);
  int col  = nt * 16 + (lane & 15);
  float v = (krow < Kreal) ? W[krow * 128 + col] : 0.0f;
  dst[t] = (__bf16)v;
}

// A fragment 16x32 bf16 (ISA 7.12.2): lane<16 -> K0..7 | K16..23 of row lane,
// lane>=16 -> K8..15 | K24..31 of row lane-16. Two ds_load_b128 per fragment.
template<int PITCH>
__device__ __forceinline__ bf16x16 load_a(const __bf16* base, int row, int kc, int lane) {
  const __bf16* p = base + row * PITCH + kc * 32 + ((lane >> 4) << 3);
  bf16x8 c1 = *(const bf16x8*)p;
  bf16x8 c2 = *(const bf16x8*)(p + 16);
  return __builtin_shufflevector(c1, c2, 0,1,2,3,4,5,6,7,8,9,10,11,12,13,14,15);
}

__device__ __forceinline__ bf16x16 load_b(const __bf16* bp, int nt) {
  bf16x8 b1 = *(const bf16x8*)(bp + nt * 512);
  bf16x8 b2 = *(const bf16x8*)(bp + nt * 512 + 8);
  return __builtin_shufflevector(b1, b2, 0,1,2,3,4,5,6,7,8,9,10,11,12,13,14,15);
}

// One 16-row M-tile against all 8 N-tiles (128 cols), K = KC*32.
// kc=0 is peeled so the first WMMA of each accumulator uses the inline-0
// C operand (no 64 zero-init v_movs). The remaining kc loop is deliberately
// NOT unrolled: keeps live B fragments bounded (<256 VGPRs, 4 waves/SIMD),
// while counted s_wait_loadcnt still pipelines loads across iterations.
template<int PITCH, int KC>
__device__ __forceinline__ void gemm_tile(f32x8 acc[8], const __bf16* aBase, int row0,
                                          const __bf16* __restrict__ wp, int lane) {
  int row = row0 + (lane & 15);
  {
    f32x8 zero;
#pragma unroll
    for (int i = 0; i < 8; ++i) zero[i] = 0.0f;
    bf16x16 a = load_a<PITCH>(aBase, row, 0, lane);
    const __bf16* bp = wp + (size_t)lane * 16;
#pragma unroll
    for (int nt = 0; nt < 8; ++nt) {
      bf16x16 b = load_b(bp, nt);
      acc[nt] = __builtin_amdgcn_wmma_f32_16x16x32_bf16(
          false, a, false, b, (short)0, zero, false, false);
    }
  }
#pragma unroll 1
  for (int kc = 1; kc < KC; ++kc) {
    bf16x16 a = load_a<PITCH>(aBase, row, kc, lane);
    const __bf16* bp = wp + (size_t)kc * 4096 + (size_t)lane * 16;
#pragma unroll
    for (int nt = 0; nt < 8; ++nt) {
      bf16x16 b = load_b(bp, nt);
      acc[nt] = __builtin_amdgcn_wmma_f32_16x16x32_bf16(
          false, a, false, b, (short)0, acc[nt], false, false);
    }
  }
}

// C/D layout: lane l, vgpr v -> row = v + 8*(l>>4), col = (l&15) + 16*nt.
// All store addresses are base + compile-time offset (v*PITCH + 16*nt).
__device__ __forceinline__ void bias_relu_to_hid(const f32x8 acc[8], const float* __restrict__ bias,
                                                 __bf16* hid, int row0, int lane) {
  int colb = lane & 15;
  int mb = row0 + ((lane >> 4) << 3);
  __bf16* base = hid + mb * HPITCH + colb;
#pragma unroll
  for (int nt = 0; nt < 8; ++nt) {
    float b = bias[colb + 16 * nt];
#pragma unroll
    for (int v = 0; v < 8; ++v) {
      float x = fmaxf(acc[nt][v] + b, 0.0f);
      base[v * HPITCH + 16 * nt] = (__bf16)x;
    }
  }
}

// bias + LayerNorm over the 128 cols of each row, done in registers.
// A row lives in one half-wave: reduce with width-16 xor shuffles.
__device__ __forceinline__ void layernorm_inplace(f32x8 acc[8], const float* __restrict__ bias,
                                                  const float* __restrict__ g,
                                                  const float* __restrict__ beta, int lane) {
  int colb = lane & 15;
#pragma unroll
  for (int nt = 0; nt < 8; ++nt) {
    float b = bias[colb + 16 * nt];
#pragma unroll
    for (int v = 0; v < 8; ++v) acc[nt][v] += b;
  }
  float s[8], s2[8];
#pragma unroll
  for (int v = 0; v < 8; ++v) { s[v] = 0.0f; s2[v] = 0.0f; }
#pragma unroll
  for (int nt = 0; nt < 8; ++nt)
#pragma unroll
    for (int v = 0; v < 8; ++v) { float x = acc[nt][v]; s[v] += x; s2[v] += x * x; }
#pragma unroll
  for (int m = 1; m < 16; m <<= 1) {
#pragma unroll
    for (int v = 0; v < 8; ++v) {
      s[v]  += __shfl_xor(s[v],  m, 16);
      s2[v] += __shfl_xor(s2[v], m, 16);
    }
  }
#pragma unroll
  for (int v = 0; v < 8; ++v) {
    float mu  = s[v] * (1.0f / 128.0f);
    float var = s2[v] * (1.0f / 128.0f) - mu * mu;
    s[v]  = mu;
    s2[v] = rsqrtf(var + LN_EPS);
  }
#pragma unroll
  for (int nt = 0; nt < 8; ++nt) {
    float gg = g[colb + 16 * nt];
    float bb = beta[colb + 16 * nt];
#pragma unroll
    for (int v = 0; v < 8; ++v)
      acc[nt][v] = (acc[nt][v] - s[v]) * s2[v] * gg + bb;
  }
}

__device__ __forceinline__ void store_msg(const f32x8 acc[8], __bf16* msg, int row0,
                                          int coloff, int lane, bool add) {
  int colb = lane & 15;
  int mb = row0 + ((lane >> 4) << 3);
  __bf16* base = msg + mb * MPITCH + coloff + colb;
#pragma unroll
  for (int nt = 0; nt < 8; ++nt)
#pragma unroll
    for (int v = 0; v < 8; ++v) {
      __bf16* p = base + v * MPITCH + 16 * nt;
      float x = acc[nt][v];
      if (add) x += (float)(*p);
      *p = (__bf16)x;
    }
}

// ---------------------------------------------------------------------------
__global__ __launch_bounds__(256, 2)
__attribute__((amdgpu_waves_per_eu(4)))
void msg_kernel(const int* __restrict__ edge_index,
                const float* __restrict__ edge_dx, const float* __restrict__ edge_attr,
                const float* __restrict__ va, const float* __restrict__ vb, const float* __restrict__ vc,
                const float* __restrict__ sv0, const float* __restrict__ sv1, const float* __restrict__ sv2,
                const float* __restrict__ sv3, const float* __restrict__ sv4, const float* __restrict__ sv5,
                const float* __restrict__ rv0, const float* __restrict__ rv1, const float* __restrict__ rv2,
                const float* __restrict__ rv3, const float* __restrict__ rv4, const float* __restrict__ rv5,
                const float* __restrict__ node_latent,
                const float* __restrict__ fe_b1, const float* __restrict__ fe_b2,
                const float* __restrict__ fe_g,  const float* __restrict__ fe_beta,
                const float* __restrict__ ee_b1, const float* __restrict__ ee_b2,
                const float* __restrict__ ee_g,  const float* __restrict__ ee_beta,
                const float* __restrict__ ie_b1, const float* __restrict__ ie_b2,
                const float* __restrict__ ie_g,  const float* __restrict__ ie_beta,
                const __bf16* __restrict__ wpack,
                float* __restrict__ out, int E)
{
  extern __shared__ __bf16 smem[];
  __bf16* sF  = smem + SF_OFF;
  __bf16* rF  = smem + RF_OFF;
  __bf16* eF  = smem + EF_OFF;
  __bf16* hid = smem + HID_OFF;
  __bf16* msg = smem + MSG_OFF;

  const int tid  = threadIdx.x;
  const int lane = tid & 31;
  const int wave = tid >> 5;
  const int e0   = blockIdx.x * 128;

  // ---- per-edge geometric features (threads 0..127, one edge each) ----
  if (tid < 128) {
    int e = e0 + tid; if (e >= E) e = E - 1;
    float a0 = va[3*e], a1 = va[3*e+1], a2 = va[3*e+2];
    float b0 = vb[3*e], b1 = vb[3*e+1], b2 = vb[3*e+2];
    float c0 = vc[3*e], c1 = vc[3*e+1], c2 = vc[3*e+2];
    const float* sp[6] = {sv0, sv1, sv2, sv3, sv4, sv5};
    const float* rp[6] = {rv0, rv1, rv2, rv3, rv4, rv5};
    const float rsgn[6] = {-1.f, -1.f, -1.f, 1.f, -1.f, 1.f};
    __bf16* srow = sF + tid * SPITCH;
    __bf16* rrow = rF + tid * SPITCH;
#pragma unroll
    for (int k = 0; k < 6; ++k) {
      float x = sp[k][3*e], y = sp[k][3*e+1], z = sp[k][3*e+2];
      srow[3*k+0] = (__bf16)(a0*x + a1*y + a2*z);
      srow[3*k+1] = (__bf16)(b0*x + b1*y + b2*z);
      srow[3*k+2] = (__bf16)(c0*x + c1*y + c2*z);
      float sg = rsgn[k];
      x = rp[k][3*e]; y = rp[k][3*e+1]; z = rp[k][3*e+2];
      rrow[3*k+0] = (__bf16)(sg*(a0*x + a1*y + a2*z));
      rrow[3*k+1] = (__bf16)(sg*(b0*x + b1*y + b2*z));
      rrow[3*k+2] = (__bf16)(sg*(c0*x + c1*y + c2*z));
    }
#pragma unroll
    for (int i = 18; i < 32; ++i) { srow[i] = (__bf16)0.0f; rrow[i] = (__bf16)0.0f; }
    __bf16* erow = eF + tid * EPITCH;
    float d0 = edge_dx[3*e], d1 = edge_dx[3*e+1], d2 = edge_dx[3*e+2];
    erow[0] = (__bf16)sqrtf(d0*d0 + d1*d1 + d2*d2);
    erow[1] = (__bf16)edge_attr[3*e];
    erow[2] = (__bf16)edge_attr[3*e+1];
    erow[3] = (__bf16)edge_attr[3*e+2];
#pragma unroll
    for (int i = 4; i < 32; ++i) erow[i] = (__bf16)0.0f;
  }
  // ---- node_latent gather+sum into msg cols [128,256): 2 threads/edge ----
  {
    int le = tid >> 1;
    int e = e0 + le; if (e >= E) e = E - 1;
    int half = tid & 1;
    int s = edge_index[e];
    int r = edge_index[E + e];
    const float4* ps = (const float4*)(node_latent + (size_t)s * 128 + half * 64);
    const float4* pr = (const float4*)(node_latent + (size_t)r * 128 + half * 64);
    __bf16* mrow = msg + le * MPITCH + 128 + half * 64;
#pragma unroll
    for (int j = 0; j < 16; ++j) {
      float4 u = ps[j];
      float4 w = pr[j];
      mrow[4*j+0] = (__bf16)(u.x + w.x);
      mrow[4*j+1] = (__bf16)(u.y + w.y);
      mrow[4*j+2] = (__bf16)(u.z + w.z);
      mrow[4*j+3] = (__bf16)(u.w + w.w);
    }
  }
  __syncthreads();  // only cross-wave sync needed; everything below is wave-local

  const int row0 = wave * 16;
  f32x8 acc[8];

  // feature encoder: senders
  gemm_tile<SPITCH, 1>(acc, sF, row0, wpack + OFF_FE_W1, lane);
  bias_relu_to_hid(acc, fe_b1, hid, row0, lane);
  gemm_tile<HPITCH, 4>(acc, hid, row0, wpack + OFF_FE_W2, lane);
  layernorm_inplace(acc, fe_b2, fe_g, fe_beta, lane);
  store_msg(acc, msg, row0, 0, lane, false);

  // feature encoder: receivers (accumulate into msg[:,0:128])
  gemm_tile<SPITCH, 1>(acc, rF, row0, wpack + OFF_FE_W1, lane);
  bias_relu_to_hid(acc, fe_b1, hid, row0, lane);
  gemm_tile<HPITCH, 4>(acc, hid, row0, wpack + OFF_FE_W2, lane);
  layernorm_inplace(acc, fe_b2, fe_g, fe_beta, lane);
  store_msg(acc, msg, row0, 0, lane, true);

  // edge encoder -> msg[:,256:384]
  gemm_tile<EPITCH, 1>(acc, eF, row0, wpack + OFF_EE_W1, lane);
  bias_relu_to_hid(acc, ee_b1, hid, row0, lane);
  gemm_tile<HPITCH, 4>(acc, hid, row0, wpack + OFF_EE_W2, lane);
  layernorm_inplace(acc, ee_b2, ee_g, ee_beta, lane);
  store_msg(acc, msg, row0, 256, lane, false);

  // interaction encoder (K=384) -> output
  gemm_tile<MPITCH, 12>(acc, msg, row0, wpack + OFF_IE_W1, lane);
  bias_relu_to_hid(acc, ie_b1, hid, row0, lane);
  gemm_tile<HPITCH, 4>(acc, hid, row0, wpack + OFF_IE_W2, lane);
  layernorm_inplace(acc, ie_b2, ie_g, ie_beta, lane);
  {
    int colb = lane & 15;
    int mb = row0 + ((lane >> 4) << 3);
#pragma unroll
    for (int nt = 0; nt < 8; ++nt)
#pragma unroll
      for (int v = 0; v < 8; ++v) {
        int e = e0 + mb + v;
        if (e < E) out[(size_t)e * 128 + colb + 16 * nt] = acc[nt][v];
      }
  }
}

// ---------------------------------------------------------------------------
extern "C" void kernel_launch(void* const* d_in, const int* in_sizes, int n_in,
                              void* d_out, int out_size, void* d_ws, size_t ws_size,
                              hipStream_t stream) {
  (void)n_in; (void)out_size; (void)ws_size;
  const int*   edge_index  = (const int*)  d_in[0];
  const float* edge_dx     = (const float*)d_in[1];
  const float* edge_attr   = (const float*)d_in[2];
  const float* va = (const float*)d_in[3];
  const float* vb = (const float*)d_in[4];
  const float* vc = (const float*)d_in[5];
  const float* sv[6]; for (int i = 0; i < 6; ++i) sv[i] = (const float*)d_in[6 + i];
  const float* rv[6]; for (int i = 0; i < 6; ++i) rv[i] = (const float*)d_in[12 + i];
  const float* node_latent = (const float*)d_in[18];
  const float* fe_W1 = (const float*)d_in[19];
  const float* fe_b1 = (const float*)d_in[20];
  const float* fe_W2 = (const float*)d_in[21];
  const float* fe_b2 = (const float*)d_in[22];
  const float* fe_g  = (const float*)d_in[23];
  const float* fe_bt = (const float*)d_in[24];
  const float* ee_W1 = (const float*)d_in[25];
  const float* ee_b1 = (const float*)d_in[26];
  const float* ee_W2 = (const float*)d_in[27];
  const float* ee_b2 = (const float*)d_in[28];
  const float* ee_g  = (const float*)d_in[29];
  const float* ee_bt = (const float*)d_in[30];
  const float* ie_W1 = (const float*)d_in[31];
  const float* ie_b1 = (const float*)d_in[32];
  const float* ie_W2 = (const float*)d_in[33];
  const float* ie_b2 = (const float*)d_in[34];
  const float* ie_g  = (const float*)d_in[35];
  const float* ie_bt = (const float*)d_in[36];

  const int E = in_sizes[0] / 2;
  __bf16* wpack = (__bf16*)d_ws;

  // pack fp32 weights -> bf16 WMMA B-fragments in workspace (runs every launch)
  auto pack = [&](const float* W, int off, int Kreal, int kchunks) {
    int total = kchunks * 4096;
    pack_weights<<<(total + 255) / 256, 256, 0, stream>>>(W, wpack + off, Kreal, kchunks);
  };
  pack(fe_W1, OFF_FE_W1, 18,  1);
  pack(fe_W2, OFF_FE_W2, 128, 4);
  pack(ee_W1, OFF_EE_W1, 4,   1);
  pack(ee_W2, OFF_EE_W2, 128, 4);
  pack(ie_W1, OFF_IE_W1, 384, 12);
  pack(ie_W2, OFF_IE_W2, 128, 4);

  const size_t smem_bytes = (size_t)SMEM_HALVES * sizeof(__bf16);
  (void)hipFuncSetAttribute((const void*)msg_kernel,
                            hipFuncAttributeMaxDynamicSharedMemorySize, (int)smem_bytes);
  const int blocks = (E + 127) / 128;
  msg_kernel<<<blocks, 256, smem_bytes, stream>>>(
      edge_index, edge_dx, edge_attr, va, vb, vc,
      sv[0], sv[1], sv[2], sv[3], sv[4], sv[5],
      rv[0], rv[1], rv[2], rv[3], rv[4], rv[5],
      node_latent,
      fe_b1, fe_b2, fe_g, fe_bt,
      ee_b1, ee_b2, ee_g, ee_bt,
      ie_b1, ie_b2, ie_g, ie_bt,
      wpack, (float*)d_out, E);
}
